// MHA_52810917871708
// MI455X (gfx1250) — matrix operand order repaired and checked by reading, not compiled, not run on hardware
//
#include <hip/hip_runtime.h>
#include <hip/hip_bf16.h>
#include <math.h>

// ---------------------------------------------------------------------------
// CDNA5 (gfx1250) RFA/Linformer attention.
// All GEMMs run on V_WMMA_F32_16X16X32_F16 (8192 MAC/instr, ~8x the fp32 WMMA
// rate) with f32 accumulation. fp32 inputs are converted to f16 once; every
// B operand is kept transposed (Bt[N][K]) so each lane's WMMA B fragment is a
// pair of 16-byte runs. phi_k/phi_q live only in LDS. HBM traffic ~100MB ->
// ~5us roofline at 23.3 TB/s; f16 WMMA keeps compute below that.
// ---------------------------------------------------------------------------

typedef _Float16 f16;
typedef __attribute__((ext_vector_type(8)))  _Float16 v8h;
typedef __attribute__((ext_vector_type(16))) _Float16 v16h;
typedef __attribute__((ext_vector_type(8)))  float    v8f;

__device__ __forceinline__ v8f wmma32(v16h a, v16h b, v8f c) {
  return __builtin_amdgcn_wmma_f32_16x16x32_f16(
      /*neg_a=*/false, a, /*neg_b=*/false, b,
      /*c_mod=*/(short)0, c, /*reuse_a=*/false, /*reuse_b=*/false);
}
__device__ __forceinline__ v16h cat8(v8h lo, v8h hi) {
  return __builtin_shufflevector(lo, hi, 0, 1, 2, 3, 4, 5, 6, 7,
                                 8, 9, 10, 11, 12, 13, 14, 15);
}
// A fragment (16x32, row-major A): lane row = lane&15, group g = lane>>4.
//   halves 0..7  = A[row][k0 + 8g .. +7]     (v0..v3)
//   halves 8..15 = A[row][k0 + 16 + 8g ..+7] (v4..v7)
__device__ __forceinline__ v16h ld_a(const f16* Ar, int k0, int g) {
  v8h lo = *(const v8h*)(Ar + k0 + g * 8);
  v8h hi = *(const v8h*)(Ar + k0 + 16 + g * 8);
  return cat8(lo, hi);
}
// B fragment (32x16) from transposed storage Bt[N][K]: lane col = lane&15,
// halves 0..7 = Bt[col][k0 + 8g ..+7], halves 8..15 = Bt[col][k0 + 16 + 8g ..+7].
__device__ __forceinline__ v16h ld_b(const f16* Btr, int k0, int g) {
  v8h lo = *(const v8h*)(Btr + k0 + g * 8);
  v8h hi = *(const v8h*)(Btr + k0 + 16 + g * 8);
  return cat8(lo, hi);
}

// ---------------------------------------------------------------------------
// Elementwise fp32 -> f16.
// ---------------------------------------------------------------------------
__global__ __launch_bounds__(256)
void cvt_f16(const float* __restrict__ src, f16* __restrict__ dst, long n) {
  const long i = (long)blockIdx.x * 256 + threadIdx.x;
  if (i < n) dst[i] = (f16)src[i];
}

// ---------------------------------------------------------------------------
// Transpose + convert: dst[b][N][M] (f16) = src[b][M][N] (f32), 32x32 tiles.
// ---------------------------------------------------------------------------
__global__ __launch_bounds__(256)
void cvt_t_f16(const float* __restrict__ src, long sSb, int ldsrc,
               f16* __restrict__ dst, long sDb, int lddst) {
  __shared__ f16 t[32][33];
  const int bm = blockIdx.y * 32, bn = blockIdx.x * 32;
  src += (long)blockIdx.z * sSb;
  dst += (long)blockIdx.z * sDb;
  const int tx = threadIdx.x & 31, tg = threadIdx.x >> 5;
  for (int r = tg; r < 32; r += 8)
    t[r][tx] = (f16)src[(long)(bm + r) * ldsrc + bn + tx];
  __syncthreads();
  for (int r = tg; r < 32; r += 8)
    dst[(long)(bn + r) * lddst + bm + tx] = t[tx][r];
}

// ---------------------------------------------------------------------------
// Generic batched GEMM: C[b](MxN,f16) = A[b](MxK,f16) * Bt[b](NxK,f16)^T.
// 256 threads; one 64x64 C tile per block, 2 16x16 sub-tiles per wave,
// K-loop in steps of 32 via v_wmma_f32_16x16x32_f16.
// ---------------------------------------------------------------------------
__global__ __launch_bounds__(256)
void gemm_f16(const f16* __restrict__ A, long sAb, int lda,
              const f16* __restrict__ Bt, long sBb, int ldb,
              f16* __restrict__ C, long sCb, int ldc, int K) {
  const int tid = threadIdx.x;
  const int lane = tid & 31, wave = tid >> 5;
  const int m15 = lane & 15, g = lane >> 4;
  const int nb = blockIdx.x * 64, mb = blockIdx.y * 64;
  A += (long)blockIdx.z * sAb;
  Bt += (long)blockIdx.z * sBb;
  C += (long)blockIdx.z * sCb;

  for (int t = 0; t < 2; ++t) {
    const int sub = wave * 2 + t;
    const int sm = mb + (sub >> 2) * 16;
    const int sn = nb + (sub & 3) * 16;
    v8f acc = {};
    const f16* Ar = A + (long)(sm + m15) * lda;
    const f16* Btr = Bt + (long)(sn + m15) * ldb;
    for (int k0 = 0; k0 < K; k0 += 32)
      acc = wmma32(ld_a(Ar, k0, g), ld_b(Btr, k0, g), acc);
    f16* Cp = C + (long)(sm + 8 * g) * ldc + sn + m15;
#pragma unroll
    for (int i = 0; i < 8; ++i) Cp[(long)i * ldc] = (f16)acc[i];
  }
}

// ---------------------------------------------------------------------------
// KV pass: one workgroup (8 waves) per (b, n).
//   xk  = normalize_rows(kh[b,:,n,:])  (A-layout LDS)
//   xvt = vh^T                          (Bt-layout LDS)
//   p = xk @ rm[n]^T  -> phi^T (sin/cos) in LDS (A-layout for s-GEMM)
//   s^T tile = (vh^T @ phi)  staged in LDS, stored transposed st[b,n,h,d]
//   z[d] = column sums of phi
// ---------------------------------------------------------------------------
__global__ __launch_bounds__(256)
void rfa_kv(const f16* __restrict__ kh, const f16* __restrict__ vh,
            const f16* __restrict__ rm, f16* __restrict__ st_ws,
            float* __restrict__ z_ws) {
  const int n = blockIdx.x, b = blockIdx.y;
  __shared__ __align__(16) f16 xk[64][72];       // normalized kh rows [k][h]
  __shared__ __align__(16) f16 xvt[64][72];      // vh transposed   [h][k]
  __shared__ __align__(16) f16 pht[2][64][72];   // phi^T           [mat][j][k]
  __shared__ __align__(16) f16 stl[64][72];      // s^T tile        [h][j]
  const int tid = threadIdx.x;
  const int lane = tid & 31, wave = tid >> 5;
  const int m15 = lane & 15, g = lane >> 4;

  // Stage kh (A-layout) and vh (transposed) from f16 global; 16 halves/thread.
  {
    const int r = tid >> 2, c = tid & 3;
    const f16* khp = kh + (long)b * 64 * 1024 + (long)r * 1024 + n * 64 + c * 16;
    const f16* vhp = vh + (long)b * 64 * 1024 + (long)r * 1024 + n * 64 + c * 16;
    *(v8h*)(&xk[r][c * 16]) = *(const v8h*)(khp);
    *(v8h*)(&xk[r][c * 16 + 8]) = *(const v8h*)(khp + 8);
    v8h vlo = *(const v8h*)(vhp);
    v8h vhi = *(const v8h*)(vhp + 8);
#pragma unroll
    for (int i = 0; i < 8; ++i) xvt[c * 16 + i][r] = vlo[i];
#pragma unroll
    for (int i = 0; i < 8; ++i) xvt[c * 16 + 8 + i][r] = vhi[i];
  }
  __syncthreads();
  if (tid < 64) {  // row L2 norm: x / (||x|| + 1e-3)
    float ss = 0.f;
    for (int h = 0; h < 64; ++h) { const float v = (float)xk[tid][h]; ss += v * v; }
    const float inv = 1.0f / (sqrtf(ss) + 0.001f);
    for (int h = 0; h < 64; ++h) xk[tid][h] = (f16)((float)xk[tid][h] * inv);
  }
  __syncthreads();

  const f16* rmn = rm + (long)n * 1024 * 64;
  const long snbase = (long)(b * 16 + n) * 2048;   // z index base
  const long hbase = (long)(b * 16 + n) * 64;      // s^T row base

  for (int jt = 0; jt < 16; ++jt) {
    const int jb = jt * 64;
    // p tile [64k x 64j] = xk(64x64) @ rm_tile^T; 2 sub-tiles per wave.
    for (int t = 0; t < 2; ++t) {
      const int sub = wave * 2 + t;
      const int sm = (sub >> 2) * 16, sn = (sub & 3) * 16;
      v8f acc = {};
      const f16* Ar = &xk[sm + m15][0];
      const f16* Btr = rmn + (long)(jb + sn + m15) * 64;  // Bt[j][h] = rm[j][h]
      acc = wmma32(ld_a(Ar, 0, g), ld_b(Btr, 0, g), acc);
      acc = wmma32(ld_a(Ar, 32, g), ld_b(Btr, 32, g), acc);
#pragma unroll
      for (int i = 0; i < 8; ++i) {   // frag (row k, col j) -> phi^T[j][k]
        const float p = acc[i];
        pht[0][sn + m15][sm + i + 8 * g] = (f16)(0.1f * __sinf(p));
        pht[1][sn + m15][sm + i + 8 * g] = (f16)(0.1f * __cosf(p));
      }
    }
    __syncthreads();
    // z[d] = sum_k phi[k][j] = row sums of phi^T
    if (tid < 128) {
      const int mat = tid >> 6, j = tid & 63;
      float ssum = 0.f;
      for (int k = 0; k < 64; ++k) ssum += (float)pht[mat][j][k];
      z_ws[snbase + mat * 1024 + jb + j] = ssum;
    }
    // s^T[h][j] = sum_k vh[k][h] * phi[k][j]:  A = phi^T rows j?? No:
    // C tile is s^T so rows = j? We store transposed via stl: compute
    // s[j][h] = phi^T(jxK=k) @ vh(kxh) with A=pht, Bt=xvt; write frag
    // transposed into stl[h][j]; then copy stl rows (contig j) to st_ws.
    for (int mat = 0; mat < 2; ++mat) {
      for (int t = 0; t < 2; ++t) {
        const int sub = wave * 2 + t;
        const int sm = (sub >> 2) * 16, sn = (sub & 3) * 16;  // sm: j, sn: h
        v8f acc = {};
        const f16* Ar = &pht[mat][sm + m15][0];
        const f16* Btr = &xvt[sn + m15][0];
        acc = wmma32(ld_a(Ar, 0, g), ld_b(Btr, 0, g), acc);
        acc = wmma32(ld_a(Ar, 32, g), ld_b(Btr, 32, g), acc);
#pragma unroll
        for (int i = 0; i < 8; ++i)          // (row j, col h) -> stl[h][j]
          stl[sn + m15][sm + i + 8 * g] = (f16)acc[i];
      }
      __syncthreads();
      {  // coalesced copy stl -> st_ws[b,n, h, mat*1024 + jb + j]
        const int r = tid >> 2, c = tid & 3;
        f16* dp = st_ws + (hbase + r) * 2048 + mat * 1024 + jb + c * 16;
        *(v8h*)(dp) = *(const v8h*)(&stl[r][c * 16]);
        *(v8h*)(dp + 8) = *(const v8h*)(&stl[r][c * 16 + 8]);
      }
      __syncthreads();
    }
  }
}

// ---------------------------------------------------------------------------
// Query pass: one workgroup per (b, n, 32-row q tile). qs accumulators live in
// WMMA registers across the whole PROJ loop; phi_q only in LDS.
// ---------------------------------------------------------------------------
__global__ __launch_bounds__(256)
void rfa_q(const f16* __restrict__ qp, const f16* __restrict__ rm,
           const f16* __restrict__ st_ws, const float* __restrict__ z_ws,
           float* __restrict__ out) {
  const int qt = blockIdx.x, n = blockIdx.y, b = blockIdx.z;
  __shared__ __align__(16) f16 xq[32][72];       // normalized q rows [m][h]
  __shared__ __align__(16) f16 ph[2][32][72];    // phi_q row-major [m][j]
  __shared__ float qzs[32];
  const int tid = threadIdx.x;
  const int lane = tid & 31, wave = tid >> 5;
  const int m15 = lane & 15, g = lane >> 4;
  const int qbase = qt * 32;

  if (tid < 128) {
    const int r = tid >> 2, c = tid & 3;
    const f16* qpp = qp + ((long)b * 512 + qbase + r) * 1024 + n * 64 + c * 16;
    *(v8h*)(&xq[r][c * 16]) = *(const v8h*)(qpp);
    *(v8h*)(&xq[r][c * 16 + 8]) = *(const v8h*)(qpp + 8);
  }
  if (tid < 32) qzs[tid] = 0.f;
  __syncthreads();
  if (tid < 32) {
    float ss = 0.f;
    for (int h = 0; h < 64; ++h) { const float v = (float)xq[tid][h]; ss += v * v; }
    const float inv = 1.0f / (sqrtf(ss) + 0.001f);
    for (int h = 0; h < 64; ++h) xq[tid][h] = (f16)((float)xq[tid][h] * inv);
  }
  __syncthreads();

  const f16* rmn = rm + (long)n * 1024 * 64;
  const long snbase = (long)(b * 16 + n) * 2048;
  const long hbase = (long)(b * 16 + n) * 64;
  const int sm = (wave >> 2) * 16;  // persistent output tile rows (q)
  const int sn = (wave & 3) * 16;   // cols (h for qs, j for p)
  v8f acc = {};                     // qs tile, carried across PROJ loop

  for (int jt = 0; jt < 16; ++jt) {
    const int jb = jt * 64;
    {  // p tile [32 x 64]: 8 sub-tiles, one per wave; K=64 -> 2 wmma
      v8f pacc = {};
      const f16* Ar = &xq[sm + m15][0];
      const f16* Btr = rmn + (long)(jb + sn + m15) * 64;
      pacc = wmma32(ld_a(Ar, 0, g), ld_b(Btr, 0, g), pacc);
      pacc = wmma32(ld_a(Ar, 32, g), ld_b(Btr, 32, g), pacc);
#pragma unroll
      for (int i = 0; i < 8; ++i) {  // (row m, col j) -> ph[m][j] (A-layout)
        const float p = pacc[i];
        ph[0][sm + i + 8 * g][sn + m15] = (f16)(0.1f * __sinf(p));
        ph[1][sm + i + 8 * g][sn + m15] = (f16)(0.1f * __cosf(p));
      }
    }
    __syncthreads();
    if (tid < 32) {  // qz[m] += phi[m,:] . z
      const float* zz = z_ws + snbase + jb;
      float ssum = 0.f;
      for (int j = 0; j < 64; ++j)
        ssum += (float)ph[0][tid][j] * zz[j] + (float)ph[1][tid][j] * zz[1024 + j];
      qzs[tid] += ssum;
    }
    // qs[m,h] += phi[m,j] * s[d(j),h]; B read from transposed s^T (contig).
#pragma unroll
    for (int mat = 0; mat < 2; ++mat) {
      const f16* Ar = &ph[mat][sm + m15][0];
      const f16* Btr = st_ws + (hbase + sn + m15) * 2048 + mat * 1024 + jb;
      acc = wmma32(ld_a(Ar, 0, g), ld_b(Btr, 0, g), acc);
      acc = wmma32(ld_a(Ar, 32, g), ld_b(Btr, 32, g), acc);
    }
    __syncthreads();  // protect ph / qzs for next iteration
  }

#pragma unroll
  for (int i = 0; i < 8; ++i) {
    const int m = sm + i + 8 * g;
    const float denom = fminf(fmaxf(fabsf(qzs[m]), 1.0f), 1e10f);
    out[((long)(b * 512 + qbase + m)) * 1024 + n * 64 + sn + m15] =
        acc[i] / denom;
  }
}

// ---------------------------------------------------------------------------
// Orchestration. Workspace: 1MB f32 z | f16 regions (~77MB total).
// ---------------------------------------------------------------------------
extern "C" void kernel_launch(void* const* d_in, const int* in_sizes, int n_in,
                              void* d_out, int out_size, void* d_ws, size_t ws_size,
                              hipStream_t stream) {
  const float* query   = (const float*)d_in[0];
  const float* key     = (const float*)d_in[1];
  const float* value   = (const float*)d_in[2];
  const float* Wq      = (const float*)d_in[3];
  const float* Wk      = (const float*)d_in[4];
  const float* Wv      = (const float*)d_in[5];
  const float* Wk_down = (const float*)d_in[6];
  const float* Wv_down = (const float*)d_in[7];
  const float* rm      = (const float*)d_in[8];  // TAU == 1.0
  float* out = (float*)d_out;

  float* z_ws = (float*)d_ws;                    // 8*16*2048 = 262144 f32
  f16* h = (f16*)(z_ws + 262144);
  f16* qx16   = h; h += (size_t)8 * 512 * 1024;   // query f16 (A)
  f16* keyT16 = h; h += (size_t)8 * 1024 * 512;   // key^T  (Bt)
  f16* valT16 = h; h += (size_t)8 * 1024 * 512;   // value^T (Bt)
  f16* WqT16  = h; h += (size_t)1024 * 1024;      // Wq^T (Bt)
  f16* WkT16  = h; h += (size_t)1024 * 1024;
  f16* WvT16  = h; h += (size_t)1024 * 1024;
  f16* Wkd16  = h; h += (size_t)64 * 512;         // Wk_down (A)
  f16* Wvd16  = h; h += (size_t)64 * 512;
  f16* rm16   = h; h += (size_t)16 * 1024 * 64;   // rm (already Bt layout)
  f16* kds16  = h; h += (size_t)8 * 64 * 1024;
  f16* vds16  = h; h += (size_t)8 * 64 * 1024;
  f16* kh16   = h; h += (size_t)8 * 64 * 1024;
  f16* vh16   = h; h += (size_t)8 * 64 * 1024;
  f16* qp16   = h; h += (size_t)8 * 512 * 1024;   // projected q (A)
  f16* st16   = h;                                // s^T: [b,n][64 h][2048 d]

  const dim3 blk(256);
  // --- f32 -> f16 conversions ---
  cvt_f16<<<dim3(16384), blk, 0, stream>>>(query, qx16, (long)8 * 512 * 1024);
  cvt_f16<<<dim3(128), blk, 0, stream>>>(Wk_down, Wkd16, (long)64 * 512);
  cvt_f16<<<dim3(128), blk, 0, stream>>>(Wv_down, Wvd16, (long)64 * 512);
  cvt_f16<<<dim3(4096), blk, 0, stream>>>(rm, rm16, (long)16 * 1024 * 64);
  // transposed conversions (Bt operands)
  cvt_t_f16<<<dim3(32, 16, 8), blk, 0, stream>>>(
      key, (long)512 * 1024, 1024, keyT16, (long)1024 * 512, 512);
  cvt_t_f16<<<dim3(32, 16, 8), blk, 0, stream>>>(
      value, (long)512 * 1024, 1024, valT16, (long)1024 * 512, 512);
  cvt_t_f16<<<dim3(32, 32, 1), blk, 0, stream>>>(Wq, 0L, 1024, WqT16, 0L, 1024);
  cvt_t_f16<<<dim3(32, 32, 1), blk, 0, stream>>>(Wk, 0L, 1024, WkT16, 0L, 1024);
  cvt_t_f16<<<dim3(32, 32, 1), blk, 0, stream>>>(Wv, 0L, 1024, WvT16, 0L, 1024);

  // --- GEMMs (f16 WMMA, f32 accumulate) ---
  // k_ds[b] = Wk_down @ key[b]   (64x1024, K=512)
  gemm_f16<<<dim3(16, 1, 8), blk, 0, stream>>>(
      Wkd16, 0L, 512, keyT16, (long)1024 * 512, 512,
      kds16, (long)64 * 1024, 1024, 512);
  gemm_f16<<<dim3(16, 1, 8), blk, 0, stream>>>(
      Wvd16, 0L, 512, valT16, (long)1024 * 512, 512,
      vds16, (long)64 * 1024, 1024, 512);
  // kh[b] = k_ds[b] @ Wk   (64x1024, K=1024)
  gemm_f16<<<dim3(16, 1, 8), blk, 0, stream>>>(
      kds16, (long)64 * 1024, 1024, WkT16, 0L, 1024,
      kh16, (long)64 * 1024, 1024, 1024);
  gemm_f16<<<dim3(16, 1, 8), blk, 0, stream>>>(
      vds16, (long)64 * 1024, 1024, WvT16, 0L, 1024,
      vh16, (long)64 * 1024, 1024, 1024);
  // q[b] = query[b] @ Wq   (512x1024, K=1024)
  gemm_f16<<<dim3(16, 8, 8), blk, 0, stream>>>(
      qx16, (long)512 * 1024, 1024, WqT16, 0L, 1024,
      qp16, (long)512 * 1024, 1024, 1024);

  // --- RFA passes ---
  rfa_kv<<<dim3(16, 8), blk, 0, stream>>>(kh16, vh16, rm16, st16, z_ws);
  rfa_q<<<dim3(16, 16, 8), blk, 0, stream>>>(qp16, rm16, st16, z_ws, out);
}
